// VisionMamba_77266461655133
// MI455X (gfx1250) — compile-verified, matrix operand order checked by reading, not run on hardware
//
#include <hip/hip_runtime.h>
#include <math.h>

typedef __attribute__((ext_vector_type(16))) _Float16 v16h;
typedef __attribute__((ext_vector_type(8)))  float    v8f;

#define Bn   2
#define Tn   8
#define NPc  196
#define Lc   1569          // T*NP + 1
#define Dm   192
#define Din  384
#define Dst  16
#define DTR  12
#define BLc  (Bn*Lc)       // 3138
#define EPSc 1e-5f

// ---------------------------------------------------------------------------
// Patch embedding + position/temporal embedding -> seq; also initializes h.
// grid (Lc, Bn), block 192
// ---------------------------------------------------------------------------
__global__ void embed_kernel(const float* __restrict__ x,
                             const float* __restrict__ pw,
                             const float* __restrict__ pb,
                             const float* __restrict__ cls,
                             const float* __restrict__ pos,
                             const float* __restrict__ tpos,
                             float* __restrict__ seq,
                             float* __restrict__ h) {
  int s = blockIdx.x, b = blockIdx.y, d = threadIdx.x;
  float val;
  if (s == 0) {
    val = cls[d] + pos[d];
  } else {
    int q = s - 1;
    int t = q / NPc, p = q % NPc;
    int ph = p / 14, pwi = p % 14;
    float acc = 0.f;
    const float* wrow = pw + (size_t)d * 768;
    for (int c = 0; c < 3; ++c) {
      for (int i = 0; i < 16; ++i) {
        const float* xr = x + ((((size_t)b*3 + c)*Tn + t)*224 + (ph*16 + i))*224 + pwi*16;
        const float* wr = wrow + (c*16 + i)*16;
#pragma unroll
        for (int j = 0; j < 16; ++j) acc = fmaf(xr[j], wr[j], acc);
      }
    }
    val = acc + pb[d] + pos[(size_t)(1 + p)*Dm + d] + tpos[(size_t)t*Dm + d];
  }
  size_t o = ((size_t)b*Lc + s)*Dm + d;
  seq[o] = val;
  h[o]   = val;
}

// ---------------------------------------------------------------------------
// res += h; rn = rmsnorm(res) * nw     grid BLc, block 256
// ---------------------------------------------------------------------------
__global__ void addnorm_kernel(float* __restrict__ res,
                               const float* __restrict__ h,
                               const float* __restrict__ nw,
                               float* __restrict__ rn) {
  int row = blockIdx.x, tid = threadIdx.x;
  __shared__ float red[256];
  float v = 0.f;
  size_t o = (size_t)row*Dm + tid;
  if (tid < Dm) { v = res[o] + h[o]; res[o] = v; }
  red[tid] = v * v;
  __syncthreads();
  for (int st = 128; st > 0; st >>= 1) {
    if (tid < st) red[tid] += red[tid + st];
    __syncthreads();
  }
  if (tid < Dm) {
    float sc = rsqrtf(red[0] / (float)Dm + EPSc);
    rn[o] = v * sc * nw[tid];
  }
}

// ---------------------------------------------------------------------------
// WMMA fragment helpers. Element i of the v16h maps to K = k0 + 8*half + i
// (i<8) and K = k0 + 16 + 8*half + (i-8), per the ISA 16-bit 16x32 layout.
// ---------------------------------------------------------------------------
__device__ __forceinline__ v16h pack16(float4 a0, float4 a1, float4 a2, float4 a3) {
  v16h r;
  r[0]=(_Float16)a0.x; r[1]=(_Float16)a0.y; r[2]=(_Float16)a0.z; r[3]=(_Float16)a0.w;
  r[4]=(_Float16)a1.x; r[5]=(_Float16)a1.y; r[6]=(_Float16)a1.z; r[7]=(_Float16)a1.w;
  r[8]=(_Float16)a2.x; r[9]=(_Float16)a2.y; r[10]=(_Float16)a2.z; r[11]=(_Float16)a2.w;
  r[12]=(_Float16)a3.x; r[13]=(_Float16)a3.y; r[14]=(_Float16)a3.z; r[15]=(_Float16)a3.w;
  return r;
}

__device__ __forceinline__ v16h load_frag(const float* __restrict__ p,
                                          int k0, int half, bool valid) {
  if (valid) {
    const float4* q0 = (const float4*)(p + k0 + 8*half);
    const float4* q1 = (const float4*)(p + k0 + 16 + 8*half);
    return pack16(q0[0], q0[1], q1[0], q1[1]);
  }
  v16h z;
#pragma unroll
  for (int i = 0; i < 16; ++i) z[i] = (_Float16)0.f;
  return z;
}

__device__ __forceinline__ v16h load_frag2(const float* __restrict__ p,
                                           const float* __restrict__ p2,
                                           int k0, int half, bool valid) {
  if (valid) {
    const float4* q0 = (const float4*)(p + k0 + 8*half);
    const float4* q1 = (const float4*)(p + k0 + 16 + 8*half);
    const float4* r0 = (const float4*)(p2 + k0 + 8*half);
    const float4* r1 = (const float4*)(p2 + k0 + 16 + 8*half);
    float4 a0 = q0[0], a1 = q0[1], a2 = q1[0], a3 = q1[1];
    float4 b0 = r0[0], b1 = r0[1], b2 = r1[0], b3 = r1[1];
    a0.x += b0.x; a0.y += b0.y; a0.z += b0.z; a0.w += b0.w;
    a1.x += b1.x; a1.y += b1.y; a1.z += b1.z; a1.w += b1.w;
    a2.x += b2.x; a2.y += b2.y; a2.z += b2.z; a2.w += b2.w;
    a3.x += b3.x; a3.y += b3.y; a3.z += b3.z; a3.w += b3.w;
    return pack16(a0, a1, a2, a3);
  }
  v16h z;
#pragma unroll
  for (int i = 0; i < 16; ++i) z[i] = (_Float16)0.f;
  return z;
}

#define WMMA_F16(Aop, Bop, Cop) \
  __builtin_amdgcn_wmma_f32_16x16x32_f16(false, (Aop), false, (Bop), (short)0, (Cop), false, false)

// ---------------------------------------------------------------------------
// Generic GEMM: C[M x Npad] = (A [+ A2]) [M x K, lda] * W[N x K]^T
// One wave per 16x64 output block (4 N-tiles), A fragment reused 4x.
// grid(ceil(M/16), ceil(Npad/64)), block 32. K must be a multiple of 32.
// ---------------------------------------------------------------------------
__global__ void wmma_gemm_kernel(const float* __restrict__ A,
                                 const float* __restrict__ A2,
                                 int lda,
                                 const float* __restrict__ W,
                                 float* __restrict__ C, int ldc,
                                 int M, int N, int Npad, int K) {
  int mt = blockIdx.x, ng = blockIdx.y;
  int lane = threadIdx.x & 31;
  int idx  = lane & 15;
  int half = lane >> 4;

  int arow   = mt*16 + idx;
  bool avalid = arow < M;
  bool fullA  = ((mt + 1) * 16) <= M;
  const float* ap  = A + (size_t)arow * lda;
  const float* ap2 = A2 ? (A2 + (size_t)arow * lda) : (const float*)0;

  int ntiles = Npad >> 4;
  int nt0 = ng * 4;
  int nsub = ntiles - nt0; if (nsub > 4) nsub = 4;

  const float* w0 = W + (size_t)((nt0 + 0)*16 + idx) * K;
  const float* w1 = W + (size_t)((nt0 + 1)*16 + idx) * K;
  const float* w2 = W + (size_t)((nt0 + 2)*16 + idx) * K;
  const float* w3 = W + (size_t)((nt0 + 3)*16 + idx) * K;
  bool v0 = (0 < nsub) && ((nt0 + 0)*16 + idx < N);
  bool v1 = (1 < nsub) && ((nt0 + 1)*16 + idx < N);
  bool v2 = (2 < nsub) && ((nt0 + 2)*16 + idx < N);
  bool v3 = (3 < nsub) && ((nt0 + 3)*16 + idx < N);

  // warm L2 with the (heavily reused) weight rows
  __builtin_prefetch(w0, 0, 1);
  if (v3) __builtin_prefetch(w3, 0, 1);

  v8f acc0 = {}, acc1 = {}, acc2 = {}, acc3 = {};

  bool allfull = fullA && (Npad == N) && (nsub == 4);
  if (allfull) {
    if (ap2) {
      for (int k0 = 0; k0 < K; k0 += 32) {
        v16h af = load_frag2(ap, ap2, k0, half, true);
        acc0 = WMMA_F16(af, load_frag(w0, k0, half, true), acc0);
        acc1 = WMMA_F16(af, load_frag(w1, k0, half, true), acc1);
        acc2 = WMMA_F16(af, load_frag(w2, k0, half, true), acc2);
        acc3 = WMMA_F16(af, load_frag(w3, k0, half, true), acc3);
      }
    } else {
      for (int k0 = 0; k0 < K; k0 += 32) {
        v16h af = load_frag(ap, k0, half, true);
        acc0 = WMMA_F16(af, load_frag(w0, k0, half, true), acc0);
        acc1 = WMMA_F16(af, load_frag(w1, k0, half, true), acc1);
        acc2 = WMMA_F16(af, load_frag(w2, k0, half, true), acc2);
        acc3 = WMMA_F16(af, load_frag(w3, k0, half, true), acc3);
      }
    }
  } else {
    for (int k0 = 0; k0 < K; k0 += 32) {
      v16h af = ap2 ? load_frag2(ap, ap2, k0, half, avalid)
                    : load_frag(ap, k0, half, avalid);
      acc0 = WMMA_F16(af, load_frag(w0, k0, half, v0), acc0);
      acc1 = WMMA_F16(af, load_frag(w1, k0, half, v1), acc1);
      acc2 = WMMA_F16(af, load_frag(w2, k0, half, v2), acc2);
      acc3 = WMMA_F16(af, load_frag(w3, k0, half, v3), acc3);
    }
  }

  int nbase = nt0*16 + idx;
#pragma unroll
  for (int r = 0; r < 8; ++r) {
    int m = mt*16 + r + 8*half;
    if (m < M) {
      float* crow = C + (size_t)m * ldc;
      if (0 < nsub) crow[nbase]      = acc0[r];
      if (1 < nsub) crow[nbase + 16] = acc1[r];
      if (2 < nsub) crow[nbase + 32] = acc2[r];
      if (3 < nsub) crow[nbase + 48] = acc3[r];
    }
  }
}

// ---------------------------------------------------------------------------
// Depthwise causal conv (k=4) + SiLU; rev=1 operates on reversed sequence and
// stores in reversed order. u = xz[..., :384] (row stride 768).
// ---------------------------------------------------------------------------
__global__ void conv_silu_kernel(const float* __restrict__ xz,
                                 const float* __restrict__ cw,
                                 const float* __restrict__ cb,
                                 float* __restrict__ uc, int rev) {
  int gid = blockIdx.x * blockDim.x + threadIdx.x;
  if (gid >= BLc * Din) return;
  int d = gid % Din;
  int i = (gid / Din) % Lc;
  int b = gid / (Din * Lc);
  float acc = cb[d];
#pragma unroll
  for (int k = 0; k < 4; ++k) {
    int j = i - 3 + k;
    if (j >= 0) {
      int l = rev ? (Lc - 1 - j) : j;
      acc = fmaf(cw[d*4 + k], xz[((size_t)b*Lc + l)*(2*Din) + d], acc);
    }
  }
  float sv = acc / (1.f + __expf(-acc));
  uc[((size_t)b*Lc + i)*Din + d] = sv;
}

// ---------------------------------------------------------------------------
// dt = softplus(dtl @ dw.T + db), dtl = xdbl[..., :12] (row stride 48)
// ---------------------------------------------------------------------------
__global__ void dt_kernel(const float* __restrict__ xdbl,
                          const float* __restrict__ dw,
                          const float* __restrict__ db,
                          float* __restrict__ dt) {
  int gid = blockIdx.x * blockDim.x + threadIdx.x;
  if (gid >= BLc * Din) return;
  int d  = gid % Din;
  int bl = gid / Din;
  float acc = db[d];
  const float* xr = xdbl + (size_t)bl * 48;
  const float* wr = dw + (size_t)d * DTR;
#pragma unroll
  for (int r = 0; r < DTR; ++r) acc = fmaf(xr[r], wr[r], acc);
  dt[(size_t)bl*Din + d] = (acc > 20.f) ? acc : log1pf(__expf(acc));
}

// ---------------------------------------------------------------------------
// Selective scan, both directions (blockIdx.y = dir). Lane layout: 16 lanes
// per (b,d) pair, one state each; cross-state reduce via shfl_xor.
// grid (48, 2), block 256  -> 16 lanes * 768 (b,d) pairs per dir.
// ---------------------------------------------------------------------------
__global__ void scan_kernel(const float* __restrict__ ucf, const float* __restrict__ dtf,
                            const float* __restrict__ xdf, const float* __restrict__ Alf,
                            const float* __restrict__ Dpf,
                            const float* __restrict__ ucb, const float* __restrict__ dtb,
                            const float* __restrict__ xdb, const float* __restrict__ Alb,
                            const float* __restrict__ Dpb,
                            const float* __restrict__ xz,
                            float* __restrict__ yf, float* __restrict__ yb) {
  int dir = blockIdx.y;
  const float* uc = dir ? ucb : ucf;
  const float* dt = dir ? dtb : dtf;
  const float* xd = dir ? xdb : xdf;
  const float* Al = dir ? Alb : Alf;
  const float* Dp = dir ? Dpb : Dpf;
  float* yout = dir ? yb : yf;

  int tid = blockIdx.x * blockDim.x + threadIdx.x;   // 0..12287
  int s  = tid & 15;
  int bd = tid >> 4;                                  // 0..767
  int d = bd % Din;
  int b = bd / Din;

  float Ac  = -__expf(Al[(size_t)d*Dst + s]);
  float Dpv = Dp[d];
  float hst = 0.f;

  for (int i = 0; i < Lc; ++i) {
    size_t row = (size_t)b*Lc + i;
    float dtv = dt[row*Din + d];
    float uv  = uc[row*Din + d];
    const float* xr = xd + row*48;
    float Bv = xr[DTR + s];
    float Cv = xr[DTR + Dst + s];
    float dA = __expf(dtv * Ac);
    hst = hst * dA + dtv * uv * Bv;
    float part = hst * Cv;
    part += __shfl_xor(part, 1, 32);
    part += __shfl_xor(part, 2, 32);
    part += __shfl_xor(part, 4, 32);
    part += __shfl_xor(part, 8, 32);
    if (s == 0) {
      int l = dir ? (Lc - 1 - i) : i;
      float zv = xz[((size_t)b*Lc + l)*(2*Din) + Din + d];
      float gate = zv / (1.f + __expf(-zv));
      yout[((size_t)b*Lc + l)*Din + d] = (part + uv * Dpv) * gate;
    }
  }
}

// ---------------------------------------------------------------------------
// Final: rmsnorm(h+res, norm_f_w) on cls tokens only, then head.
// grid 1, block 64
// ---------------------------------------------------------------------------
__global__ void head_kernel(const float* __restrict__ h, const float* __restrict__ res,
                            const float* __restrict__ nfw,
                            const float* __restrict__ hw, const float* __restrict__ hb,
                            float* __restrict__ out) {
  __shared__ float v[Bn][Dm];
  __shared__ float scale[Bn];
  int tid = threadIdx.x;
  for (int b = 0; b < Bn; ++b)
    for (int d = tid; d < Dm; d += blockDim.x) {
      size_t o = ((size_t)b*Lc)*Dm + d;
      v[b][d] = h[o] + res[o];
    }
  __syncthreads();
  if (tid < Bn) {
    float ss = 0.f;
    for (int d = 0; d < Dm; ++d) { float t = v[tid][d]; ss += t*t; }
    scale[tid] = rsqrtf(ss / (float)Dm + EPSc);
  }
  __syncthreads();
  for (int idx = tid; idx < Bn*22; idx += blockDim.x) {
    int b = idx / 22, c = idx % 22;
    float acc = 0.f;
    for (int d = 0; d < Dm; ++d)
      acc = fmaf(v[b][d] * scale[b] * nfw[d], hw[(size_t)c*Dm + d], acc);
    out[idx] = acc + hb[c];
  }
}

// ---------------------------------------------------------------------------
extern "C" void kernel_launch(void* const* d_in, const int* in_sizes, int n_in,
                              void* d_out, int out_size, void* d_ws, size_t ws_size,
                              hipStream_t stream) {
  (void)in_sizes; (void)n_in; (void)out_size; (void)ws_size;
  const float* x      = (const float*)d_in[0];
  const float* patchw = (const float*)d_in[1];
  const float* patchb = (const float*)d_in[2];
  const float* cls    = (const float*)d_in[3];
  const float* pos    = (const float*)d_in[4];
  const float* tpos   = (const float*)d_in[5];
  const float* normws = (const float*)d_in[6];
  const float* iws    = (const float*)d_in[7];
  const float* cwf    = (const float*)d_in[8];
  const float* cbf    = (const float*)d_in[9];
  const float* xwf    = (const float*)d_in[10];
  const float* dwf    = (const float*)d_in[11];
  const float* dbf    = (const float*)d_in[12];
  const float* Alf    = (const float*)d_in[13];
  const float* Dpf    = (const float*)d_in[14];
  const float* cwb    = (const float*)d_in[15];
  const float* cbb    = (const float*)d_in[16];
  const float* xwb    = (const float*)d_in[17];
  const float* dwb    = (const float*)d_in[18];
  const float* dbb    = (const float*)d_in[19];
  const float* Alb    = (const float*)d_in[20];
  const float* Dpb    = (const float*)d_in[21];
  const float* ows    = (const float*)d_in[22];
  const float* nfw    = (const float*)d_in[23];
  const float* hw     = (const float*)d_in[24];
  const float* hb     = (const float*)d_in[25];
  float* out = (float*)d_out;

  // workspace carve-up
  float* ws  = (float*)d_ws;
  float* res = ws;             ws += (size_t)BLc*Dm;
  float* h   = ws;             ws += (size_t)BLc*Dm;
  float* rn  = ws;             ws += (size_t)BLc*Dm;
  float* xz  = ws;             ws += (size_t)BLc*(2*Din);
  float* ucf = ws;             ws += (size_t)BLc*Din;
  float* ucb = ws;             ws += (size_t)BLc*Din;
  float* xdf = ws;             ws += (size_t)BLc*48;
  float* xdb = ws;             ws += (size_t)BLc*48;
  float* dtf = ws;             ws += (size_t)BLc*Din;
  float* dtb = ws;             ws += (size_t)BLc*Din;
  float* yfb = ws;             ws += (size_t)BLc*Din;
  float* ybb = ws;             ws += (size_t)BLc*Din;

  const int MT = (BLc + 15) / 16;                 // 197 M-tiles
  const int CB = (BLc*Din + 255) / 256;           // elementwise blocks

  embed_kernel<<<dim3(Lc, Bn), 192, 0, stream>>>(x, patchw, patchb, cls, pos, tpos, res, h);

  for (int layer = 0; layer < 24; ++layer) {
    addnorm_kernel<<<BLc, 256, 0, stream>>>(res, h, normws + (size_t)layer*Dm, rn);

    // xz = rn @ in_proj.T   [BLc x 192] x [768 x 192]^T  -> 12 groups of 4 N-tiles
    wmma_gemm_kernel<<<dim3(MT, 12), 32, 0, stream>>>(
        rn, (const float*)0, Dm, iws + (size_t)layer*2*Din*Dm, xz, 2*Din,
        BLc, 2*Din, 2*Din, Dm);

    conv_silu_kernel<<<CB, 256, 0, stream>>>(xz, cwf + (size_t)layer*Din*4,
                                             cbf + (size_t)layer*Din, ucf, 0);
    conv_silu_kernel<<<CB, 256, 0, stream>>>(xz, cwb + (size_t)layer*Din*4,
                                             cbb + (size_t)layer*Din, ucb, 1);

    // xdbl = uc @ x_proj.T  [BLc x 384] x [44 x 384]^T  (N padded to 48)
    wmma_gemm_kernel<<<dim3(MT, 1), 32, 0, stream>>>(
        ucf, (const float*)0, Din, xwf + (size_t)layer*44*Din, xdf, 48,
        BLc, 44, 48, Din);
    wmma_gemm_kernel<<<dim3(MT, 1), 32, 0, stream>>>(
        ucb, (const float*)0, Din, xwb + (size_t)layer*44*Din, xdb, 48,
        BLc, 44, 48, Din);

    dt_kernel<<<CB, 256, 0, stream>>>(xdf, dwf + (size_t)layer*Din*DTR,
                                      dbf + (size_t)layer*Din, dtf);
    dt_kernel<<<CB, 256, 0, stream>>>(xdb, dwb + (size_t)layer*Din*DTR,
                                      dbb + (size_t)layer*Din, dtb);

    scan_kernel<<<dim3(48, 2), 256, 0, stream>>>(
        ucf, dtf, xdf, Alf + (size_t)layer*Din*Dst, Dpf + (size_t)layer*Din,
        ucb, dtb, xdb, Alb + (size_t)layer*Din*Dst, Dpb + (size_t)layer*Din,
        xz, yfb, ybb);

    // h = (yf + yb) @ out_proj.T  [BLc x 384] x [192 x 384]^T -> 3 groups of 4
    wmma_gemm_kernel<<<dim3(MT, 3), 32, 0, stream>>>(
        yfb, ybb, Din, ows + (size_t)layer*Dm*Din, h, Dm,
        BLc, Dm, Dm, Din);
  }

  head_kernel<<<1, 64, 0, stream>>>(h, res, nfw, hw, hb, out);
}
